// TemporalTransformerBlock_40492951666717
// MI455X (gfx1250) — matrix operand order repaired and checked by reading, not compiled, not run on hardware
//
#include <hip/hip_runtime.h>
#include <math.h>

// Problem constants (match reference)
#define BATCH 2048
#define TT    16
#define DIMC  512
#define NH    8
#define HD    64
#define FFIN  2048   // INNER
#define NREL  31
#define GB    4      // batch rows per block (weight-reuse factor)

typedef __attribute__((ext_vector_type(16))) __bf16 v16bf;
typedef __attribute__((ext_vector_type(8)))  float  v8f;

__device__ __forceinline__ v8f wmma_bf16(v16bf a, v16bf b, v8f c) {
  // D = A(16x32 bf16) * B(32x16 bf16) + C(16x16 f32)
  return __builtin_amdgcn_wmma_f32_16x16x32_bf16(false, a, false, b, (short)0, c,
                                                 false, false);
}

// A fragment: 16x32 bf16 tile from row-major src[16][stride], k-offset k0.
// lane m=L&15; element j -> K = 16*(L>>4) + j (contiguous, matches load_bT_frag).
__device__ __forceinline__ v16bf load_a_frag(const __bf16* src, int stride, int k0, int lane) {
  int m  = lane & 15;
  int kg = (lane >> 4) * 16;
  const __bf16* p = src + (size_t)m * stride + k0;
  v16bf a;
#pragma unroll
  for (int j = 0; j < 8; ++j) a[j] = p[kg + j];
#pragma unroll
  for (int j = 0; j < 8; ++j) a[8 + j] = p[kg + 8 + j];
  return a;
}

// B fragment: B[k][n] tile supplied via transposed storage bt[n][k] (row stride = K-extent).
// Lane n=L&15, halves j -> K = 16*(L>>4) + j  => contiguous 32B per lane.
__device__ __forceinline__ v16bf load_bT_frag(const __bf16* bt, int stride, int n0, int k0, int lane) {
  int n  = lane & 15;
  int kb = (lane >> 4) * 16;
  const __bf16* p = bt + (size_t)(n0 + n) * stride + k0 + kb;
  v16bf b;
#pragma unroll
  for (int j = 0; j < 16; ++j) b[j] = p[j];
  return b;
}

// ---------------- prep kernels ----------------

// out[c][r] = bf16(in[r][c])   (store W^T in bf16)
__global__ void transpose_bf16_kernel(const float* __restrict__ in, __bf16* __restrict__ out,
                                      int R, int C) {
  size_t idx = (size_t)blockIdx.x * blockDim.x + threadIdx.x;
  if (idx < (size_t)R * C) {
    int r = (int)(idx / C), c = (int)(idx % C);
    out[(size_t)c * R + r] = (__bf16)in[idx];
  }
}

// out[r][c] = bf16( pe[r,:] @ wmat[:,c] ), r in [0,32) with row 31 zero-padded.
__global__ void peproj_kernel(const float* __restrict__ pe, const float* __restrict__ wmat,
                              __bf16* __restrict__ out) {
  int r = blockIdx.x;   // 0..31
  int c = threadIdx.x;  // 0..511
  float acc = 0.f;
  if (r < NREL) {
    for (int k = 0; k < DIMC; ++k)
      acc += pe[(size_t)r * DIMC + k] * wmat[(size_t)k * DIMC + c];
  }
  out[(size_t)r * DIMC + c] = (__bf16)acc;
}

// shared LayerNorm helper: wave handles GB rows (row = w + 16*g), writes bf16 rows to xn
__device__ __forceinline__ void layernorm_rows(const float* __restrict__ xb, __bf16* xn,
                                               const float* __restrict__ lng,
                                               const float* __restrict__ lnb,
                                               int w, int lane) {
#pragma unroll
  for (int g = 0; g < GB; ++g) {
    int row = w + 16 * g;
    const float* xr = xb + (size_t)row * DIMC;
    float s = 0.f, s2 = 0.f;
#pragma unroll
    for (int i = 0; i < 16; ++i) {
      float v = xr[i * 32 + lane];
      s += v; s2 += v * v;
    }
#pragma unroll
    for (int off = 16; off >= 1; off >>= 1) {
      s  += __shfl_xor(s, off, 32);
      s2 += __shfl_xor(s2, off, 32);
    }
    float mean = s * (1.f / DIMC);
    float var  = s2 * (1.f / DIMC) - mean * mean;
    float rs   = rsqrtf(var + 1e-5f);
#pragma unroll
    for (int i = 0; i < 16; ++i) {
      int c = i * 32 + lane;
      xn[(size_t)row * DIMC + c] = (__bf16)((xr[c] - mean) * rs * lng[c] + lnb[c]);
    }
  }
}

// ---------------- fused temporal attention: GB batch rows per block ----------------
// LDS: xn 64x512 bf16 (64KB, reused for attn-out) | q,k,v 64x512 bf16 (3x64KB)
//      simb [GB][8][16][16] f32 (32KB) | atb same bf16 (16KB)  => 304KB total

__global__ __launch_bounds__(512) void attn_kernel(
    const float* __restrict__ xin, float* __restrict__ xout,
    const float* __restrict__ lng, const float* __restrict__ lnb,
    const __bf16* __restrict__ wqT, const __bf16* __restrict__ wkT,
    const __bf16* __restrict__ wvT, const __bf16* __restrict__ woT,
    const float* __restrict__ bo,
    const __bf16* __restrict__ krr, const __bf16* __restrict__ qrr) {
  extern __shared__ char smem[];
  __bf16* xn   = (__bf16*)(smem);             // 64x512; later reused as attn-out
  __bf16* qs   = (__bf16*)(smem + 65536);
  __bf16* ks   = (__bf16*)(smem + 131072);
  __bf16* vs   = (__bf16*)(smem + 196608);
  float*  simb = (float*)(smem + 262144);     // [GB*8][16][16]
  __bf16* atb  = (__bf16*)(smem + 294912);    // [GB*8][16][16]

  const int tid  = threadIdx.x;
  const int lane = tid & 31;
  const int w    = tid >> 5;                  // wave 0..15
  const int b0   = blockIdx.x * GB;
  const float* xb = xin + (size_t)b0 * TT * DIMC;
  float* xo       = xout + (size_t)b0 * TT * DIMC;

  // ---- 1. LayerNorm (GB*16 rows) ----
  layernorm_rows(xb, xn, lng, lnb, w, lane);
  __syncthreads();

  // ---- 2. QKV projection: B-fragment loaded once, reused for GB row-blocks ----
  for (int i = 0; i < 6; ++i) {
    int tile = w * 6 + i;                     // 96 column-tiles (q|k|v x 32)
    int sel  = tile >> 5;
    int cb   = tile & 31;
    const __bf16* wT = (sel == 0) ? wqT : (sel == 1 ? wkT : wvT);
    __bf16* dst      = (sel == 0) ? qs  : (sel == 1 ? ks  : vs);
    v8f c[GB];
#pragma unroll
    for (int g = 0; g < GB; ++g) c[g] = (v8f){};
#pragma unroll 4
    for (int kk = 0; kk < 16; ++kk) {
      v16bf bb = load_bT_frag(wT, DIMC, cb * 16, kk * 32, lane);
#pragma unroll
      for (int g = 0; g < GB; ++g) {
        v16bf a = load_a_frag(xn + (size_t)g * 16 * DIMC, DIMC, kk * 32, lane);
        c[g] = wmma_bf16(a, bb, c[g]);
      }
    }
    int n = lane & 15, mb8 = (lane >> 4) * 8;
#pragma unroll
    for (int g = 0; g < GB; ++g)
#pragma unroll
      for (int e = 0; e < 8; ++e)
        dst[(size_t)(g * 16 + mb8 + e) * DIMC + cb * 16 + n] = (__bf16)c[g][e];
  }
  __syncthreads();

  // ---- 3. per-head scores + contextual rel-pos bias; one wave per (rb, head) ----
  int nc = lane & 15, mb8 = (lane >> 4) * 8;
#pragma unroll
  for (int it = 0; it < 2; ++it) {
    int inst = w + 16 * it;                   // 0..31
    int rb = inst >> 3, h = inst & 7;
    int k0h = h * HD;
    const __bf16* qh = qs + (size_t)rb * 16 * DIMC;
    const __bf16* kh = ks + (size_t)rb * 16 * DIMC;
    float* sim = &simb[(size_t)(rb * 8 + h) * 256];

    // S = q_h @ k_h^T
    v8f s = {};
#pragma unroll
    for (int kk = 0; kk < 2; ++kk) {
      v16bf a  = load_a_frag(qh, DIMC, k0h + kk * 32, lane);
      v16bf bb = load_bT_frag(kh, DIMC, 0, k0h + kk * 32, lane);
      s = wmma_bf16(a, bb, s);
    }
#pragma unroll
    for (int e = 0; e < 8; ++e) sim[(mb8 + e) * 16 + nc] = s[e];

    // E[m][r] = q_h[m].KrR_h[r]; F[n][r] = k_h[n].QrR_h[r]  (r=31 pad is zero)
#pragma unroll
    for (int cb2 = 0; cb2 < 2; ++cb2) {
      v8f ev = {}, fv = {};
#pragma unroll
      for (int kk = 0; kk < 2; ++kk) {
        v16bf aq = load_a_frag(qh, DIMC, k0h + kk * 32, lane);
        v16bf bk = load_bT_frag(krr, DIMC, cb2 * 16, k0h + kk * 32, lane);
        ev = wmma_bf16(aq, bk, ev);
        v16bf ak = load_a_frag(kh, DIMC, k0h + kk * 32, lane);
        v16bf bq = load_bT_frag(qrr, DIMC, cb2 * 16, k0h + kk * 32, lane);
        fv = wmma_bf16(ak, bq, fv);
      }
      int r = cb2 * 16 + nc;
      if (r < NREL) {
        // same wave owns this sim block: plain RMW, lanes hit distinct addresses
#pragma unroll
        for (int e = 0; e < 8; ++e) {         // bias1: sim[m][r+m-15] += E[m][r]
          int row = mb8 + e;
          int n = r + row - (TT - 1);
          if (n >= 0 && n < TT) sim[row * 16 + n] += ev[e];
        }
#pragma unroll
        for (int e = 0; e < 8; ++e) {         // bias2: sim[n-r+15][n] += F[n][r]
          int nrow = mb8 + e;
          int m = nrow - r + (TT - 1);
          if (m >= 0 && m < TT) sim[m * 16 + nrow] += fv[e];
        }
      }
    }
  }
  __syncthreads();

  // ---- 4. softmax (mask all-true): one thread per (rb,h,row) ----
  {
    float* sr = &simb[(size_t)(tid >> 4) * 256 + (size_t)(tid & 15) * 16];
    __bf16* ar = &atb[(size_t)(tid >> 4) * 256 + (size_t)(tid & 15) * 16];
    const float scale = 0.125f;               // 64^-0.5
    float mx = -1e30f;
#pragma unroll
    for (int n = 0; n < 16; ++n) { float v = sr[n] * scale; sr[n] = v; mx = fmaxf(mx, v); }
    float sum = 0.f;
#pragma unroll
    for (int n = 0; n < 16; ++n) { float e = __expf(sr[n] - mx); sr[n] = e; sum += e; }
    float inv = 1.f / sum;
#pragma unroll
    for (int n = 0; n < 16; ++n) ar[n] = (__bf16)(sr[n] * inv);
  }
  __syncthreads();

  // ---- 5. out = attn @ v (K=16 padded to 32); write into xn region ----
  for (int i = 0; i < 8; ++i) {
    int tile = w + 16 * i;                    // 0..127
    int rb = tile >> 5, rem = tile & 31;
    int h = rem >> 2, db = rem & 3;
    bool lo = lane < 16;
    v16bf a;
    {
      const __bf16* ap = &atb[(size_t)(rb * 8 + h) * 256 + (size_t)nc * 16];
#pragma unroll
      for (int j = 0; j < 16; ++j) a[j] = lo ? ap[j] : (__bf16)0.0f;   // K=16..31 (hi lanes) zero
    }
    v16bf bb;
#pragma unroll
    for (int j = 0; j < 16; ++j)
      bb[j] = lo ? vs[(size_t)(rb * 16 + j) * DIMC + h * HD + db * 16 + nc]
                 : (__bf16)0.0f;
    v8f c = {};
    c = wmma_bf16(a, bb, c);
#pragma unroll
    for (int e = 0; e < 8; ++e)
      xn[(size_t)(rb * 16 + mb8 + e) * DIMC + h * HD + db * 16 + nc] = (__bf16)c[e];
  }
  __syncthreads();

  // ---- 6. final: out @ wo + bo + residual; B-frag reused across GB row-blocks ----
  for (int i = 0; i < 2; ++i) {
    int oc = (w * 2 + i) * 16;
    v8f c[GB];
#pragma unroll
    for (int g = 0; g < GB; ++g) c[g] = (v8f){};
#pragma unroll 4
    for (int kk = 0; kk < 16; ++kk) {
      v16bf bb = load_bT_frag(woT, DIMC, oc, kk * 32, lane);
#pragma unroll
      for (int g = 0; g < GB; ++g) {
        v16bf a = load_a_frag(xn + (size_t)g * 16 * DIMC, DIMC, kk * 32, lane);
        c[g] = wmma_bf16(a, bb, c[g]);
      }
    }
    int col = oc + nc;
#pragma unroll
    for (int g = 0; g < GB; ++g)
#pragma unroll
      for (int e = 0; e < 8; ++e) {
        int row = g * 16 + mb8 + e;
        xo[(size_t)row * DIMC + col] = c[g][e] + bo[col] + xb[(size_t)row * DIMC + col];
      }
  }
}

// ---------------- fused GEGLU FF (in-place): GB batch rows per block ----------------
// LDS: xn 64x512 bf16 (64KB) | yb 64x256 bf16 (32KB)

__global__ __launch_bounds__(512) void ff_kernel(
    float* __restrict__ xio,
    const float* __restrict__ lng, const float* __restrict__ lnb,
    const __bf16* __restrict__ w1T, const float* __restrict__ b1,
    const __bf16* __restrict__ w2T, const float* __restrict__ b2) {
  extern __shared__ char smem[];
  __bf16* xn = (__bf16*)smem;                 // 64x512
  __bf16* yb = (__bf16*)(smem + 65536);       // 64x256

  const int tid = threadIdx.x, lane = tid & 31, w = tid >> 5;
  const int b0 = blockIdx.x * GB;
  float* xb = xio + (size_t)b0 * TT * DIMC;

  layernorm_rows(xb, xn, lng, lnb, w, lane);
  __syncthreads();

  const int nc = lane & 15, mb8 = (lane >> 4) * 8;
  v8f acc[GB][2];
#pragma unroll
  for (int g = 0; g < GB; ++g) { acc[g][0] = (v8f){}; acc[g][1] = (v8f){}; }

  for (int ch = 0; ch < 8; ++ch) {            // 8 chunks x 256 inner cols
    int colg = ch * 256 + w * 16;             // this wave's 16-col slice of INNER
    v8f av[GB], gv[GB];
#pragma unroll
    for (int g = 0; g < GB; ++g) { av[g] = (v8f){}; gv[g] = (v8f){}; }
#pragma unroll 2
    for (int kk = 0; kk < 16; ++kk) {
      v16bf b_a = load_bT_frag(w1T, DIMC, colg,        kk * 32, lane);
      v16bf b_g = load_bT_frag(w1T, DIMC, FFIN + colg, kk * 32, lane);
#pragma unroll
      for (int g = 0; g < GB; ++g) {
        v16bf a = load_a_frag(xn + (size_t)g * 16 * DIMC, DIMC, kk * 32, lane);
        av[g] = wmma_bf16(a, b_a, av[g]);
        gv[g] = wmma_bf16(a, b_g, gv[g]);
      }
    }
    float ba = b1[colg + nc], bg = b1[FFIN + colg + nc];
#pragma unroll
    for (int g = 0; g < GB; ++g)
#pragma unroll
      for (int e = 0; e < 8; ++e) {
        float aa  = av[g][e] + ba;
        float gg  = gv[g][e] + bg;
        float gel = 0.5f * gg * (1.f + erff(gg * 0.70710678118654752f));
        yb[(size_t)(g * 16 + mb8 + e) * 256 + w * 16 + nc] = (__bf16)(aa * gel);
      }
    __syncthreads();
    // out_acc += y(64x256) @ w2[chunk rows, :]; B-frag reused across row-blocks
#pragma unroll
    for (int i = 0; i < 2; ++i) {
      int oc = w * 32 + i * 16;
#pragma unroll
      for (int kk = 0; kk < 8; ++kk) {
        v16bf bb = load_bT_frag(w2T, FFIN, oc, ch * 256 + kk * 32, lane);
#pragma unroll
        for (int g = 0; g < GB; ++g) {
          v16bf a = load_a_frag(yb + (size_t)g * 16 * 256, 256, kk * 32, lane);
          acc[g][i] = wmma_bf16(a, bb, acc[g][i]);
        }
      }
    }
    __syncthreads();
  }
  // epilogue: + b2 + residual, in place (each element RMW'd by exactly one thread)
#pragma unroll
  for (int i = 0; i < 2; ++i) {
    int col = w * 32 + i * 16 + nc;
#pragma unroll
    for (int g = 0; g < GB; ++g)
#pragma unroll
      for (int e = 0; e < 8; ++e) {
        size_t idx = (size_t)(g * 16 + mb8 + e) * DIMC + col;
        xb[idx] = acc[g][i][e] + b2[col] + xb[idx];
      }
  }
}

// ---------------- host launch ----------------

extern "C" void kernel_launch(void* const* d_in, const int* in_sizes, int n_in,
                              void* d_out, int out_size, void* d_ws, size_t ws_size,
                              hipStream_t stream) {
  const float* x   = (const float*)d_in[0];
  // d_in[1] femb unused; d_in[2] attn_mask all-true (ignored)
  const float* n1g = (const float*)d_in[3];
  const float* n1b = (const float*)d_in[4];
  const float* wq1 = (const float*)d_in[5];
  const float* wk1 = (const float*)d_in[6];
  const float* wv1 = (const float*)d_in[7];
  const float* wo1 = (const float*)d_in[8];
  const float* bo1 = (const float*)d_in[9];
  const float* pe1 = (const float*)d_in[10];
  const float* n2g = (const float*)d_in[11];
  const float* n2b = (const float*)d_in[12];
  const float* wq2 = (const float*)d_in[13];
  const float* wk2 = (const float*)d_in[14];
  const float* wv2 = (const float*)d_in[15];
  const float* wo2 = (const float*)d_in[16];
  const float* bo2 = (const float*)d_in[17];
  const float* pe2 = (const float*)d_in[18];
  const float* n3g = (const float*)d_in[19];
  const float* n3b = (const float*)d_in[20];
  const float* ffw1 = (const float*)d_in[21];
  const float* ffb1 = (const float*)d_in[22];
  const float* ffw2 = (const float*)d_in[23];
  const float* ffb2 = (const float*)d_in[24];

  char* ws = (char*)d_ws;
  size_t off = 0;
  float* xmid = (float*)(ws + off); off += (size_t)BATCH * TT * DIMC * sizeof(float);
  auto alloc_bf = [&](size_t elems) { __bf16* p = (__bf16*)(ws + off); off += elems * 2; return p; };
  __bf16* wq1T = alloc_bf(512 * 512);
  __bf16* wk1T = alloc_bf(512 * 512);
  __bf16* wv1T = alloc_bf(512 * 512);
  __bf16* wo1T = alloc_bf(512 * 512);
  __bf16* wq2T = alloc_bf(512 * 512);
  __bf16* wk2T = alloc_bf(512 * 512);
  __bf16* wv2T = alloc_bf(512 * 512);
  __bf16* wo2T = alloc_bf(512 * 512);
  __bf16* w1T  = alloc_bf((size_t)512 * 4096);
  __bf16* w2T  = alloc_bf((size_t)2048 * 512);
  __bf16* krr1 = alloc_bf(32 * 512);
  __bf16* qrr1 = alloc_bf(32 * 512);
  __bf16* krr2 = alloc_bf(32 * 512);
  __bf16* qrr2 = alloc_bf(32 * 512);

  const int TPB = 256;
  auto nb = [&](size_t n) { return (int)((n + TPB - 1) / TPB); };
  transpose_bf16_kernel<<<nb(512 * 512), TPB, 0, stream>>>(wq1, wq1T, 512, 512);
  transpose_bf16_kernel<<<nb(512 * 512), TPB, 0, stream>>>(wk1, wk1T, 512, 512);
  transpose_bf16_kernel<<<nb(512 * 512), TPB, 0, stream>>>(wv1, wv1T, 512, 512);
  transpose_bf16_kernel<<<nb(512 * 512), TPB, 0, stream>>>(wo1, wo1T, 512, 512);
  transpose_bf16_kernel<<<nb(512 * 512), TPB, 0, stream>>>(wq2, wq2T, 512, 512);
  transpose_bf16_kernel<<<nb(512 * 512), TPB, 0, stream>>>(wk2, wk2T, 512, 512);
  transpose_bf16_kernel<<<nb(512 * 512), TPB, 0, stream>>>(wv2, wv2T, 512, 512);
  transpose_bf16_kernel<<<nb(512 * 512), TPB, 0, stream>>>(wo2, wo2T, 512, 512);
  transpose_bf16_kernel<<<nb((size_t)512 * 4096), TPB, 0, stream>>>(ffw1, w1T, 512, 4096);
  transpose_bf16_kernel<<<nb((size_t)2048 * 512), TPB, 0, stream>>>(ffw2, w2T, 2048, 512);

  peproj_kernel<<<32, 512, 0, stream>>>(pe1, wk1, krr1);
  peproj_kernel<<<32, 512, 0, stream>>>(pe1, wq1, qrr1);
  peproj_kernel<<<32, 512, 0, stream>>>(pe2, wk2, krr2);
  peproj_kernel<<<32, 512, 0, stream>>>(pe2, wq2, qrr2);

  const int nblk = BATCH / GB;                 // 512 blocks
  const size_t smem_attn = 311296;             // 304KB: xn + q + k + v + sim + attn
  const size_t smem_ff   = 98304;              // 96KB: xn + y chunk
  attn_kernel<<<nblk, 512, smem_attn, stream>>>(x, xmid, n1g, n1b,
                                                wq1T, wk1T, wv1T, wo1T, bo1, krr1, qrr1);
  attn_kernel<<<nblk, 512, smem_attn, stream>>>(xmid, (float*)d_out, n2g, n2b,
                                                wq2T, wk2T, wv2T, wo2T, bo2, krr2, qrr2);
  ff_kernel<<<nblk, 512, smem_ff, stream>>>((float*)d_out, n3g, n3b, w1T, ffb1, w2T, ffb2);
}
